// GatedScreeningTile_37288906064178
// MI455X (gfx1250) — matrix operand order repaired and checked by reading, not compiled, not run on hardware
//
#include <hip/hip_runtime.h>
#include <math.h>

// Problem constants (match reference)
#define B_   2
#define N_   2048
#define DIM_ 2048
#define H_   16
#define DK_  64
#define DV_  128
#define DIN_ 3072   // (DK+DV)*H

typedef __attribute__((ext_vector_type(16))) __bf16 v16bf;
typedef __attribute__((ext_vector_type(8)))  float  v8f;
typedef __attribute__((ext_vector_type(4)))  unsigned int v4u;

union Frag { v16bf bf; unsigned int u32[8]; };

__device__ __forceinline__ unsigned short f2bf(float f) {
  unsigned int u = __float_as_uint(f);
  unsigned int r = (u + 0x7FFFu + ((u >> 16) & 1u)) >> 16;  // RNE
  return (unsigned short)r;
}

__device__ __forceinline__ v8f zero8() {
  v8f z;
#pragma unroll
  for (int i = 0; i < 8; ++i) z[i] = 0.0f;
  return z;
}

// A-matrix (16x32, 16-bit) K offset for VGPR r, lane group g (contiguous: k = 16g + 2r)
__device__ __forceinline__ int a_kofs(int r, int g) {
  return g * 16 + 2 * r;
}

// Low 32 bits of a generic pointer to __shared__ == DS-space address.
__device__ __forceinline__ unsigned lds_off(const void* p) {
  return (unsigned)(size_t)p;
}

// CDNA5 async global->LDS copy, 16B per lane (tracked by ASYNCcnt).
__device__ __forceinline__ void async_copy_b128(unsigned lds_addr,
                                                const void* gaddr) {
  asm volatile("global_load_async_to_lds_b128 %0, %1, off"
               :
               : "v"(lds_addr), "v"((unsigned long long)(size_t)gaddr)
               : "memory");
}
__device__ __forceinline__ void wait_async0() {
  asm volatile("s_wait_asynccnt 0x0" ::: "memory");
}

// ---------------------------------------------------------------------------
// fp32 -> bf16 conversion
// ---------------------------------------------------------------------------
__global__ void cvt_f32_bf16(const float* __restrict__ src,
                             unsigned short* __restrict__ dst, int n) {
  int i = blockIdx.x * blockDim.x + threadIdx.x;
  int stride = gridDim.x * blockDim.x;
  for (; i < n; i += stride) dst[i] = f2bf(src[i]);
}

// ---------------------------------------------------------------------------
// bf16 GEMM: C[M,N] (f32) = A[M,K] (bf16) * B[K,N] (bf16), row-major.
// Block tile 128x128, BK=32, 256 threads = 8 waves (2 row x 4 col groups);
// each wave computes 64x32 via 4x2 WMMA 16x16x32 bf16 tiles.
// Double-buffered LDS; A staged via async global->LDS b128, B transposed
// through registers; all prefetch streams use running pointers.
// ---------------------------------------------------------------------------
__launch_bounds__(256)
__global__ void gemm_bf16(const unsigned short* __restrict__ A,
                          const unsigned short* __restrict__ Bm,
                          float* __restrict__ C, int M, int N, int K) {
  __shared__ __align__(16) unsigned short As[2][128 * 32];  // [m][k]
  __shared__ __align__(16) unsigned short Bs[2][128 * 32];  // [n][k]

  const int tid = threadIdx.x;
  const int wid = tid >> 5, lane = tid & 31;
  const int g = lane >> 4, l16 = lane & 15;
  const int warpRow = wid >> 2;   // 0..1, 64 rows each
  const int warpCol = wid & 3;    // 0..3, 32 cols each
  const int mBase = blockIdx.y * 128;
  const int nBase = blockIdx.x * 128;

  // A copy: 512 x 16B chunks (row = c>>2, k-ofs = (c&3)*8); 2 per thread
  const int am0 = tid >> 2,         ak0 = (tid & 3) * 8;
  const int am1 = (tid + 256) >> 2, ak1 = ((tid + 256) & 3) * 8;
  // B transpose copy: one 2(k) x 8(n) block per thread
  const int bk2 = (tid >> 4) * 2;   // 0..30
  const int bn8 = (tid & 15) * 8;   // 0..120

  v8f acc[4][2];
#pragma unroll
  for (int a = 0; a < 4; ++a)
#pragma unroll
    for (int bn = 0; bn < 2; ++bn) acc[a][bn] = zero8();

  const int T = K / 32;

  // Running prefetch pointers (strength-reduced addressing)
  const unsigned short* ap0 = &A[(size_t)(mBase + am0) * K + ak0];
  const unsigned short* ap1 = &A[(size_t)(mBase + am1) * K + ak1];
  const unsigned short* bp = &Bm[(size_t)bk2 * N + nBase + bn8];
  const size_t bstep = (size_t)32 * N;

  // Prologue: A(0) async -> As[0]; B(0) -> regs
  async_copy_b128(lds_off(&As[0][am0 * 32 + ak0]), ap0);
  async_copy_b128(lds_off(&As[0][am1 * 32 + ak1]), ap1);
  ap0 += 32;
  ap1 += 32;
  v4u w0 = *(const v4u*)bp;
  v4u w1 = *(const v4u*)(bp + N);
  bp += bstep;

  for (int t = 0; t < T; ++t) {
    const int cur = t & 1;

    // Stage B(t) regs -> Bs[cur] in [n][k] layout (packed b32 stores)
#pragma unroll
    for (int j = 0; j < 4; ++j) {
      unsigned lo = (w0[j] & 0xffffu) | (w1[j] << 16);
      unsigned hi = (w0[j] >> 16) | (w1[j] & 0xffff0000u);
      *(unsigned int*)&Bs[cur][(bn8 + 2 * j) * 32 + bk2] = lo;
      *(unsigned int*)&Bs[cur][(bn8 + 2 * j + 1) * 32 + bk2] = hi;
    }
    // Prefetch B(t+1) into regs (global, no LDS hazard)
    if (t + 1 < T) {
      w0 = *(const v4u*)bp;
      w1 = *(const v4u*)(bp + N);
      bp += bstep;
    }
    wait_async0();       // A(t) fully in LDS
    __syncthreads();     // B(t) stores visible; prior readers of buf cur done
    // Issue A(t+1) async into the other buffer (overlaps with compute)
    if (t + 1 < T) {
      async_copy_b128(lds_off(&As[cur ^ 1][am0 * 32 + ak0]), ap0);
      async_copy_b128(lds_off(&As[cur ^ 1][am1 * 32 + ak1]), ap1);
      ap0 += 32;
      ap1 += 32;
    }

    Frag fa[4], fb[2];
#pragma unroll
    for (int a = 0; a < 4; ++a) {
      int m = warpRow * 64 + a * 16 + l16;
#pragma unroll
      for (int r = 0; r < 8; ++r)
        fa[a].u32[r] = *(const unsigned int*)&As[cur][m * 32 + a_kofs(r, g)];
    }
#pragma unroll
    for (int bn = 0; bn < 2; ++bn) {
      int n = warpCol * 32 + bn * 16 + l16;
#pragma unroll
      for (int r = 0; r < 8; ++r)
        fb[bn].u32[r] =
            *(const unsigned int*)&Bs[cur][n * 32 + g * 16 + 2 * r];
    }
#pragma unroll
    for (int a = 0; a < 4; ++a)
#pragma unroll
      for (int bn = 0; bn < 2; ++bn)
        acc[a][bn] = __builtin_amdgcn_wmma_f32_16x16x32_bf16(
            false, fa[a].bf, false, fb[bn].bf, (short)0, acc[a][bn], false,
            false);
  }

  // C layout: VGPR r -> row r + 8*g, col = l16; pointer-bump stores
#pragma unroll
  for (int a = 0; a < 4; ++a) {
    const int row0 = mBase + warpRow * 64 + a * 16 + 8 * g;
#pragma unroll
    for (int bn = 0; bn < 2; ++bn) {
      const int col = nBase + warpCol * 32 + bn * 16 + l16;
      float* cp = C + (size_t)row0 * N + col;
#pragma unroll
      for (int r = 0; r < 8; ++r) {
        *cp = acc[a][bn][r];
        cp += N;
      }
    }
  }
}

// ---------------------------------------------------------------------------
// Split heads, L2-normalize q/k/v (-> bf16), gate factor tanh(silu(g))*e^hws.
// One block per token; wave w handles heads w and w+8.
// ---------------------------------------------------------------------------
__launch_bounds__(256)
__global__ void split_normalize(const float* __restrict__ qg,
                                const float* __restrict__ kv,
                                const float* __restrict__ ls_hws,
                                unsigned short* __restrict__ qn,
                                unsigned short* __restrict__ kn,
                                unsigned short* __restrict__ vn,
                                float* __restrict__ gatef) {
  const int t = blockIdx.x;  // 0..B*N-1
  const int b = t / N_, n = t % N_;
  const int wid = threadIdx.x >> 5, lane = threadIdx.x & 31;

  for (int hh = 0; hh < 2; ++hh) {
    const int h = wid + hh * 8;
    const float* qrow = qg + (size_t)t * DIN_ + h * (DK_ + DV_);
    const float* grow = qrow + DK_;
    const float* krow = kv + (size_t)t * DIN_ + h * (DK_ + DV_);
    const float* vrow = krow + DK_;
    const size_t hn = (size_t)(b * H_ + h) * N_ + n;

    // q (64)
    float q0 = qrow[lane], q1 = qrow[lane + 32];
    float ssq = q0 * q0 + q1 * q1;
#pragma unroll
    for (int o = 16; o > 0; o >>= 1) ssq += __shfl_xor(ssq, o, 32);
    float sc = 1.0f / fmaxf(sqrtf(ssq), 1e-12f);
    qn[hn * DK_ + lane] = f2bf(q0 * sc);
    qn[hn * DK_ + lane + 32] = f2bf(q1 * sc);

    // k (64)
    float k0 = krow[lane], k1 = krow[lane + 32];
    ssq = k0 * k0 + k1 * k1;
#pragma unroll
    for (int o = 16; o > 0; o >>= 1) ssq += __shfl_xor(ssq, o, 32);
    sc = 1.0f / fmaxf(sqrtf(ssq), 1e-12f);
    kn[hn * DK_ + lane] = f2bf(k0 * sc);
    kn[hn * DK_ + lane + 32] = f2bf(k1 * sc);

    // v (128)
    float v0 = vrow[lane], v1 = vrow[lane + 32];
    float v2 = vrow[lane + 64], v3 = vrow[lane + 96];
    ssq = v0 * v0 + v1 * v1 + v2 * v2 + v3 * v3;
#pragma unroll
    for (int o = 16; o > 0; o >>= 1) ssq += __shfl_xor(ssq, o, 32);
    sc = 1.0f / fmaxf(sqrtf(ssq), 1e-12f);
    vn[hn * DV_ + lane] = f2bf(v0 * sc);
    vn[hn * DV_ + lane + 32] = f2bf(v1 * sc);
    vn[hn * DV_ + lane + 64] = f2bf(v2 * sc);
    vn[hn * DV_ + lane + 96] = f2bf(v3 * sc);

    // gate factor (128): tanh(silu(g)) * exp(ls_hws[h])
    const float hscale = expf(ls_hws[h]);
#pragma unroll
    for (int j = 0; j < 4; ++j) {
      float gv = grow[lane + 32 * j];
      float si = gv / (1.0f + expf(-gv));
      gatef[hn * DV_ + lane + 32 * j] = tanhf(si) * hscale;
    }
  }
}

// ---------------------------------------------------------------------------
// Fused screened attention: per (b, h, 64-row i-tile).
// 8 waves = 4 row groups (16 rows) x 2 col groups (64 of DV=128).
// j streamed in 32-row tiles, double-buffered; q/k staged by async
// global->LDS b128, v transposed through registers. sim = q k^T (WMMA),
// attn = relu(1-r(1-sim))^2, restage D->A layout via private LDS,
// aggr += attn @ v (WMMA). Epilogue: tanh_norm via ds_add_f32 row
// reduction, apply gate, store bf16 to [B,N,H*DV].
// ---------------------------------------------------------------------------
__launch_bounds__(256)
__global__ void attention(const unsigned short* __restrict__ qn,
                          const unsigned short* __restrict__ kn,
                          const unsigned short* __restrict__ vn,
                          const float* __restrict__ gatef,
                          const float* __restrict__ ls_iaw,
                          unsigned short* __restrict__ attout) {
  __shared__ __align__(16) unsigned short qT[64 * 64];        // [row][k]
  __shared__ __align__(16) unsigned short kTb[2][32 * 64];    // [keyrow][k]
  __shared__ __align__(16) unsigned short vTb[2][128 * 32];   // [col][kj]
  __shared__ __align__(16) unsigned short stage[8][16 * 32];  // per-wave attn
  __shared__ float rowssq[64];

  const int iBase = blockIdx.x * 64;
  const int h = blockIdx.y, b = blockIdx.z;
  const int tid = threadIdx.x, wid = tid >> 5, lane = tid & 31;
  const int g = lane >> 4, l16 = lane & 15;
  const int rowGroup = wid & 3;   // 4 x 16 rows
  const int colGroup = wid >> 2;  // 2 x 64 cols

  const float rh = 1.0f / (expf(ls_iaw[h]) + 1.0f);

  const size_t headN = (size_t)(b * H_ + h) * N_;
  const unsigned short* qhead = qn + headN * DK_;
  const unsigned short* khead = kn + headN * DK_;
  const unsigned short* vhead = vn + headN * DV_;
  const float* ghead = gatef + headN * DV_;

  // Copy geometry: q rows are 64 ushort = 8 x 16B chunks; k rows likewise.
  const int qr0 = tid >> 3,         qk0 = (tid & 7) * 8;
  const int qr1 = (tid + 256) >> 3, qk1 = ((tid + 256) & 7) * 8;
  const int kr = tid >> 3,          kk8 = (tid & 7) * 8;  // 256 chunks
  // v transpose: one 2(kj) x 8(c) block per thread
  const int vk2 = (tid >> 4) * 2;   // 0..30
  const int vc8 = (tid & 15) * 8;   // 0..120

  // Running prefetch pointers
  const unsigned short* kp = &khead[(size_t)kr * DK_ + kk8];
  const unsigned short* vp = &vhead[(size_t)vk2 * DV_ + vc8];

  // Prologue: q tile + k(0) async; v(0) -> regs
  async_copy_b128(lds_off(&qT[qr0 * 64 + qk0]),
                  &qhead[(size_t)(iBase + qr0) * DK_ + qk0]);
  async_copy_b128(lds_off(&qT[qr1 * 64 + qk1]),
                  &qhead[(size_t)(iBase + qr1) * DK_ + qk1]);
  async_copy_b128(lds_off(&kTb[0][kr * 64 + kk8]), kp);
  kp += 32 * DK_;
  v4u w0 = *(const v4u*)vp;
  v4u w1 = *(const v4u*)(vp + DV_);
  vp += 32 * DV_;

  v8f agg[4];
#pragma unroll
  for (int i = 0; i < 4; ++i) agg[i] = zero8();

  const int T = N_ / 32;
  for (int jt = 0; jt < T; ++jt) {
    const int cur = jt & 1;

    // Stage v(jt) regs -> vTb[cur] ([col][kj], packed b32)
#pragma unroll
    for (int j = 0; j < 4; ++j) {
      unsigned lo = (w0[j] & 0xffffu) | (w1[j] << 16);
      unsigned hi = (w0[j] >> 16) | (w1[j] & 0xffff0000u);
      *(unsigned int*)&vTb[cur][(vc8 + 2 * j) * 32 + vk2] = lo;
      *(unsigned int*)&vTb[cur][(vc8 + 2 * j + 1) * 32 + vk2] = hi;
    }
    if (jt + 1 < T) {
      w0 = *(const v4u*)vp;
      w1 = *(const v4u*)(vp + DV_);
      vp += 32 * DV_;
    }
    wait_async0();    // k(jt) (and on jt==0 the q tile) fully in LDS
    __syncthreads();
    if (jt + 1 < T) {
      async_copy_b128(lds_off(&kTb[cur ^ 1][kr * 64 + kk8]), kp);
      kp += 32 * DK_;
    }

    // sim: 16 rows x 32 keys per wave (K=64 -> two WMMA K-steps)
    v8f sim[2];
    sim[0] = zero8();
    sim[1] = zero8();
#pragma unroll
    for (int ks = 0; ks < 2; ++ks) {
      int kk = ks * 32;
      Frag fa;
      int m = rowGroup * 16 + l16;
#pragma unroll
      for (int r = 0; r < 8; ++r)
        fa.u32[r] = *(const unsigned int*)&qT[m * 64 + kk + a_kofs(r, g)];
#pragma unroll
      for (int nt = 0; nt < 2; ++nt) {
        Frag fb;
        int nn = nt * 16 + l16;
#pragma unroll
        for (int r = 0; r < 8; ++r)
          fb.u32[r] =
              *(const unsigned int*)&kTb[cur][nn * 64 + kk + g * 16 + 2 * r];
        sim[nt] = __builtin_amdgcn_wmma_f32_16x16x32_bf16(
            false, fa.bf, false, fb.bf, (short)0, sim[nt], false, false);
      }
    }

    // attn = relu(1 - r*(1-sim))^2; restage D-layout -> A-layout via LDS
    unsigned short* st = &stage[wid][0];
#pragma unroll
    for (int nt = 0; nt < 2; ++nt)
#pragma unroll
      for (int r = 0; r < 8; ++r) {
        float s = sim[nt][r];
        float a = 1.0f - rh * (1.0f - s);
        a = fmaxf(a, 0.0f);
        a = a * a;
        st[(r + 8 * g) * 32 + nt * 16 + l16] = f2bf(a);
      }
    asm volatile("s_wait_dscnt 0x0" ::: "memory");  // same-wave LDS RAW

    Frag fat;
#pragma unroll
    for (int r = 0; r < 8; ++r)
      fat.u32[r] = *(const unsigned int*)&st[l16 * 32 + a_kofs(r, g)];
#pragma unroll
    for (int nt4 = 0; nt4 < 4; ++nt4) {
      Frag fv;
      int c = colGroup * 64 + nt4 * 16 + l16;
#pragma unroll
      for (int r = 0; r < 8; ++r)
        fv.u32[r] =
            *(const unsigned int*)&vTb[cur][c * 32 + g * 16 + 2 * r];
      agg[nt4] = __builtin_amdgcn_wmma_f32_16x16x32_bf16(
          false, fat.bf, false, fv.bf, (short)0, agg[nt4], false, false);
    }
  }

  // Epilogue: row sum of squares across both col groups (full DV=128)
  __syncthreads();
  if (tid < 64) rowssq[tid] = 0.0f;
  __syncthreads();
#pragma unroll
  for (int nt4 = 0; nt4 < 4; ++nt4)
#pragma unroll
    for (int r = 0; r < 8; ++r) {
      float v = agg[nt4][r];
      atomicAdd(&rowssq[rowGroup * 16 + r + 8 * g], v * v);
    }
  __syncthreads();

  float ts[8];
#pragma unroll
  for (int r = 0; r < 8; ++r) {
    float nrm = sqrtf(rowssq[rowGroup * 16 + r + 8 * g]);
    ts[r] = tanhf(nrm) / fmaxf(nrm, 1e-12f);
  }
#pragma unroll
  for (int nt4 = 0; nt4 < 4; ++nt4) {
    const int col = colGroup * 64 + nt4 * 16 + l16;
#pragma unroll
    for (int r = 0; r < 8; ++r) {
      const int nGlob = iBase + rowGroup * 16 + r + 8 * g;
      float gfac = ghead[(size_t)nGlob * DV_ + col];
      float outv = agg[nt4][r] * ts[r] * gfac;
      attout[(size_t)(b * N_ + nGlob) * (H_ * DV_) + h * DV_ + col] =
          f2bf(outv);
    }
  }
}

// ---------------------------------------------------------------------------
// Host launcher
// ---------------------------------------------------------------------------
extern "C" void kernel_launch(void* const* d_in, const int* in_sizes, int n_in,
                              void* d_out, int out_size, void* d_ws,
                              size_t ws_size, hipStream_t stream) {
  const float* tokens = (const float*)d_in[0];
  const float* Wqg = (const float*)d_in[1];
  const float* Wkv = (const float*)d_in[2];
  const float* Wo = (const float*)d_in[3];
  const float* ls_iaw = (const float*)d_in[4];
  const float* ls_hws = (const float*)d_in[5];
  float* out = (float*)d_out;

  char* ws = (char*)d_ws;
  size_t off = 0;
  auto alloc = [&](size_t bytes) -> void* {
    void* p = ws + off;
    off += (bytes + 255) & ~(size_t)255;
    return p;
  };

  const size_t nTok = (size_t)B_ * N_ * DIM_;
  const size_t nWqg = (size_t)DIM_ * DIN_;
  const size_t nWo = (size_t)(H_ * DV_) * DIM_;
  const size_t nHeadK = (size_t)B_ * H_ * N_ * DK_;
  const size_t nHeadV = (size_t)B_ * H_ * N_ * DV_;

  unsigned short* tok_bf = (unsigned short*)alloc(nTok * 2);
  unsigned short* wqg_bf = (unsigned short*)alloc(nWqg * 2);
  unsigned short* wkv_bf = (unsigned short*)alloc(nWqg * 2);
  unsigned short* wo_bf = (unsigned short*)alloc(nWo * 2);
  float* qg_f32 = (float*)alloc((size_t)B_ * N_ * DIN_ * 4);
  float* kv_f32 = (float*)alloc((size_t)B_ * N_ * DIN_ * 4);
  unsigned short* qn = (unsigned short*)alloc(nHeadK * 2);
  unsigned short* kn = (unsigned short*)alloc(nHeadK * 2);
  unsigned short* vn = (unsigned short*)alloc(nHeadV * 2);
  float* gatef = (float*)alloc(nHeadV * 4);
  unsigned short* attout = (unsigned short*)alloc(nTok * 2);

  // 1) convert inputs to bf16
  cvt_f32_bf16<<<dim3((unsigned)((nTok + 255) / 256)), 256, 0, stream>>>(
      tokens, tok_bf, (int)nTok);
  cvt_f32_bf16<<<dim3((unsigned)((nWqg + 255) / 256)), 256, 0, stream>>>(
      Wqg, wqg_bf, (int)nWqg);
  cvt_f32_bf16<<<dim3((unsigned)((nWqg + 255) / 256)), 256, 0, stream>>>(
      Wkv, wkv_bf, (int)nWqg);
  cvt_f32_bf16<<<dim3((unsigned)((nWo + 255) / 256)), 256, 0, stream>>>(
      Wo, wo_bf, (int)nWo);

  // 2) qg / kv projections
  gemm_bf16<<<dim3(DIN_ / 128, (B_ * N_) / 128), 256, 0, stream>>>(
      tok_bf, wqg_bf, qg_f32, B_ * N_, DIN_, DIM_);
  gemm_bf16<<<dim3(DIN_ / 128, (B_ * N_) / 128), 256, 0, stream>>>(
      tok_bf, wkv_bf, kv_f32, B_ * N_, DIN_, DIM_);

  // 3) split heads + normalize + gate factor
  split_normalize<<<dim3(B_ * N_), 256, 0, stream>>>(qg_f32, kv_f32, ls_hws,
                                                     qn, kn, vn, gatef);

  // 4) fused screened attention + tanh_norm + gating
  attention<<<dim3(N_ / 64, H_, B_), 256, 0, stream>>>(qn, kn, vn, gatef,
                                                       ls_iaw, attout);

  // 5) output projection
  gemm_bf16<<<dim3(DIM_ / 128, (B_ * N_) / 128), 256, 0, stream>>>(
      attout, wo_bf, out, B_ * N_, DIM_, H_ * DV_);
}
